// VectorQuantizer_60601988547210
// MI455X (gfx1250) — matrix operand order repaired and checked by reading, not compiled, not run on hardware
//
#include <hip/hip_runtime.h>
#include <hip/hip_bf16.h>

typedef __attribute__((ext_vector_type(16))) __bf16 v16bf;
typedef __attribute__((ext_vector_type(8)))  float  v8f;
typedef __attribute__((ext_vector_type(4)))  float  f32x4;
typedef __attribute__((ext_vector_type(2)))  float  f32x2;

#define N_TOKENS  131072
#define EMBED_DIM 64
#define NUM_CODES 1024

// ---------------------------------------------------------------------------
// Kernel 1: normalize codebook rows -> bf16, zero histogram + SSE accumulator
// One wave (32 lanes) per codebook row; lane holds 2 floats.
// ---------------------------------------------------------------------------
__global__ void vq_prep(const float* __restrict__ weight,
                        unsigned short* __restrict__ ncb_u16,
                        unsigned* __restrict__ counts,
                        unsigned long long* __restrict__ sse) {
    int tid = blockIdx.x * blockDim.x + threadIdx.x;
    if (tid < NUM_CODES) counts[tid] = 0u;
    if (tid == 0) *sse = 0ull;

    int row  = tid >> 5;          // 1024 waves total (128 blocks x 8 waves)
    int lane = threadIdx.x & 31;
    if (row < NUM_CODES) {
        const f32x2* wr = (const f32x2*)(weight + (size_t)row * EMBED_DIM);
        f32x2 w = wr[lane];
        float s = w[0]*w[0] + w[1]*w[1];
        #pragma unroll
        for (int m = 1; m < 32; m <<= 1) s += __shfl_xor(s, m, 32);
        float scale = 1.0f / fmaxf(sqrtf(s), 1e-12f);
        __bf16* orow = ((__bf16*)ncb_u16) + (size_t)row * EMBED_DIM + lane * 2;
        orow[0] = (__bf16)(w[0] * scale);
        orow[1] = (__bf16)(w[1] * scale);
    }
}

// ---------------------------------------------------------------------------
// Kernel 2: fused GEMM(argmax) + gather + loss partials.
// One wave computes a 16-token x 1024-code similarity via 128 bf16 WMMAs.
//
// A layout (16-bit A 16x32, ISA 7.12.2): lane l -> row m = l&15;
//   half = l>>4: loaded K-contiguous, v[0..3] = K = 16*half+0..7, v[4..7] = K = 16*half+8..15.
// B layout (16-bit B 32x16): lane l -> col n = l&15; half selects K 0-15/16-31;
//   VGPR j packs k = 16*half + {2j,2j+1}  ==> lane reads 16 *contiguous* bf16
//   from normalized-codebook row n (since B[k][n] = ncb[n][k]).
// C/D layout: lane l, vgpr j -> S[m = j + 8*(l>>4)][n = l&15].
//
// Inside the code-tile loop, each lane's candidate code index is strictly
// increasing, so a strict ">" keeps the lowest index on ties for free; the
// index tie-break is only needed in the cross-lane reduction.
// ---------------------------------------------------------------------------
__global__ void vq_assign(const float* __restrict__ x,
                          const float* __restrict__ weight,
                          const unsigned short* __restrict__ ncb_u16,
                          float* __restrict__ qout,        // d_out + 1
                          int* __restrict__ idx_out,
                          unsigned* __restrict__ counts,
                          unsigned long long* __restrict__ sse) {
    const __bf16* ncb = (const __bf16*)ncb_u16;
    const int lane = threadIdx.x & 31;
    const int wave = threadIdx.x >> 5;
    const int tile = blockIdx.x * (blockDim.x >> 5) + wave;   // 16-token tile
    const int tok0 = tile * 16;
    const int m    = lane & 15;
    const int half = lane >> 4;

    // Load A tile (16 tokens x 64 dims) into WMMA bf16 layout, both K-steps.
    const float* arow = x + (size_t)(tok0 + m) * EMBED_DIM;
    v16bf a[2];
    #pragma unroll
    for (int s = 0; s < 2; ++s) {
        const int kb = s * 32 + half * 16;
        #pragma unroll
        for (int e = 0; e < 8; ++e) {
            a[s][e]     = (__bf16)arow[kb + e];
            a[s][8 + e] = (__bf16)arow[kb + 8 + e];
        }
    }

    float bestv[8];
    int   besti[8];
    #pragma unroll
    for (int j = 0; j < 8; ++j) { bestv[j] = -1e30f; besti[j] = 0; }

    #pragma unroll 4
    for (int c = 0; c < NUM_CODES / 16; ++c) {
        // B: contiguous 32B reads from normalized codebook row (c*16 + m).
        const __bf16* brow = ncb + (size_t)(c * 16 + m) * EMBED_DIM + half * 16;
        v16bf b0 = *(const v16bf*)(brow);        // K-step 0
        v16bf b1 = *(const v16bf*)(brow + 32);   // K-step 1
        v8f acc = {};
        acc = __builtin_amdgcn_wmma_f32_16x16x32_bf16(false, a[0], false, b0,
                                                      (short)0, acc, false, false);
        acc = __builtin_amdgcn_wmma_f32_16x16x32_bf16(false, a[1], false, b1,
                                                      (short)0, acc, false, false);
        const int code = c * 16 + m;
        #pragma unroll
        for (int j = 0; j < 8; ++j) {
            float v = acc[j];
            if (v > bestv[j]) { bestv[j] = v; besti[j] = code; }  // strict >
        }
    }

    // Cross-lane argmax across the 16 columns held within each half
    // (index tie-break needed here: lanes hold different code columns).
    #pragma unroll
    for (int msk = 1; msk < 16; msk <<= 1) {
        #pragma unroll
        for (int j = 0; j < 8; ++j) {
            float ov = __shfl_xor(bestv[j], msk, 32);
            int   oi = __shfl_xor(besti[j], msk, 32);
            if (ov > bestv[j] || (ov == bestv[j] && oi < besti[j])) {
                bestv[j] = ov; besti[j] = oi;
            }
        }
    }

    // Writer lanes 0 (rows 0..7) and 16 (rows 8..15): indices + histogram.
    if (m == 0) {
        #pragma unroll
        for (int j = 0; j < 8; ++j) {
            idx_out[tok0 + half * 8 + j] = besti[j];
            atomicAdd(&counts[besti[j]], 1u);
        }
    }

    // Gather quantized rows, straight-through output, SSE partial.
    float lsse = 0.f;
    #pragma unroll 1
    for (int t = 0; t < 16; ++t) {
        int id = __shfl(besti[t & 7], (t >> 3) * 16, 32);   // token t = j + 8*half
        f32x2 q  = ((const f32x2*)(weight + (size_t)id * EMBED_DIM))[lane];
        f32x2 xv = ((const f32x2*)(x + (size_t)(tok0 + t) * EMBED_DIM))[lane];
        f32x2 d  = { q[0] - xv[0], q[1] - xv[1] };
        lsse += d[0]*d[0] + d[1]*d[1];
        f32x2 st = { xv[0] + d[0], xv[1] + d[1] };          // straight-through
        // write-once stream: keep it out of L2's way
        __builtin_nontemporal_store(
            st, (f32x2*)(qout + (size_t)(tok0 + t) * EMBED_DIM) + lane);
    }
    #pragma unroll
    for (int msk = 1; msk < 32; msk <<= 1) lsse += __shfl_xor(lsse, msk, 32);
    if (lane == 0)  // fixed-point accumulate -> deterministic across replays
        atomicAdd(sse, (unsigned long long)(lsse * 1024.0f + 0.5f));
}

// ---------------------------------------------------------------------------
// Kernel 3: one-hot encodings [N, 1024]. Wave per row; 512B coalesced
// non-temporal b128 stores (512 MB write-once stream; bypass L2 retention).
// ---------------------------------------------------------------------------
__global__ void vq_encodings(const int* __restrict__ idx,
                             float* __restrict__ enc) {
    int lane = threadIdx.x & 31;
    int wave = threadIdx.x >> 5;
    int n = blockIdx.x * (blockDim.x >> 5) + wave;
    int id = idx[n];
    float* row = enc + (size_t)n * NUM_CODES;
    #pragma unroll
    for (int it = 0; it < 8; ++it) {
        int col = it * 128 + lane * 4;
        f32x4 v = {};
        if (id >= col && id < col + 4) v[id - col] = 1.0f;
        __builtin_nontemporal_store(v, (f32x4*)(row + col));
    }
}

// ---------------------------------------------------------------------------
// Kernel 4: perplexity + loss.
// ---------------------------------------------------------------------------
__global__ void vq_finalize(const unsigned* __restrict__ counts,
                            const unsigned long long* __restrict__ sse,
                            float* __restrict__ loss_out,
                            float* __restrict__ perp_out) {
    __shared__ float red[32];
    int tid = threadIdx.x;
    float p = (float)counts[tid] * (1.0f / (float)N_TOKENS);
    float h = p * logf(p + 1e-10f);
    #pragma unroll
    for (int m = 1; m < 32; m <<= 1) h += __shfl_xor(h, m, 32);
    if ((tid & 31) == 0) red[tid >> 5] = h;
    __syncthreads();
    if (tid < 32) {
        float v = red[tid];
        #pragma unroll
        for (int m = 1; m < 32; m <<= 1) v += __shfl_xor(v, m, 32);
        if (tid == 0) {
            *perp_out = expf(-v);
            double s = (double)(*sse) * (1.0 / 1024.0);
            *loss_out = (float)(1.25 * s / (double)((size_t)N_TOKENS * EMBED_DIM));
        }
    }
}

// ---------------------------------------------------------------------------
// d_out layout (f32): [0]=loss, [1..1+N*D)=quantized_st, [1+N*D]=perplexity,
//                     [2+N*D ..) = encodings [N, 1024].
// ---------------------------------------------------------------------------
extern "C" void kernel_launch(void* const* d_in, const int* in_sizes, int n_in,
                              void* d_out, int out_size, void* d_ws, size_t ws_size,
                              hipStream_t stream) {
    const float* x = (const float*)d_in[0];   // [131072, 64]
    const float* w = (const float*)d_in[1];   // [1024, 64]
    float* out = (float*)d_out;
    char* ws = (char*)d_ws;

    unsigned short*     ncb    = (unsigned short*)(ws);                       // 128 KB bf16
    int*                idx    = (int*)(ws + 131072);                         // 512 KB
    unsigned*           counts = (unsigned*)(ws + 131072 + 524288);           //   4 KB
    unsigned long long* sse    = (unsigned long long*)(ws + 131072 + 524288 + 4096);

    float* loss = out;
    float* qst  = out + 1;
    float* perp = out + 1 + (size_t)N_TOKENS * EMBED_DIM;
    float* enc  = out + 2 + (size_t)N_TOKENS * EMBED_DIM;

    vq_prep     <<<128,            256, 0, stream>>>(w, ncb, counts, sse);
    vq_assign   <<<N_TOKENS / 128, 256, 0, stream>>>(x, w, ncb, qst, idx, counts, sse);
    vq_encodings<<<N_TOKENS / 8,   256, 0, stream>>>(idx, enc);
    vq_finalize <<<1,             1024, 0, stream>>>(counts, sse, loss, perp);
}
